// BahdanauAttention_84215718740129
// MI455X (gfx1250) — matrix operand order repaired and checked by reading, not compiled, not run on hardware
//
#include <hip/hip_runtime.h>

typedef __bf16 bf16_t;
typedef __attribute__((ext_vector_type(4)))  __bf16 v4bf;
typedef __attribute__((ext_vector_type(8)))  __bf16 v8bf;
typedef __attribute__((ext_vector_type(16))) __bf16 v16bf;
typedef __attribute__((ext_vector_type(8)))  float  v8f;

#define B_  128
#define T_  576
#define D_  1024
#define H_  1024
#define U_  512

#define TB    64            // t-rows per block in the score kernel
#define APAD  8
#define ASTR  (D_ + APAD)   // 1032 bf16 per LDS row (bank spread)

// Fast tanh: prefer native gfx1250 V_TANH_F32; else branch-free exp-based.
#if __has_builtin(__builtin_amdgcn_tanhf)
__device__ __forceinline__ float fast_tanh(float x) { return __builtin_amdgcn_tanhf(x); }
#elif __has_builtin(__builtin_amdgcn_tanh_f32)
__device__ __forceinline__ float fast_tanh(float x) { return __builtin_amdgcn_tanh_f32(x); }
#else
__device__ __forceinline__ float fast_tanh(float x) {
    // tanh(x) = 1 - 2/(exp(2x)+1); inputs here are O(1), no range issues
    float e = __expf(2.0f * x);
    return 1.0f - 2.0f / (e + 1.0f);
}
#endif

// ---------------- K0: W1 [D,U] f32 -> W1t [U,D] bf16 (transpose + convert) ----
__global__ void k0_w1_to_bf16t(const float* __restrict__ W1, bf16_t* __restrict__ W1t) {
    int i = blockIdx.x * 256 + threadIdx.x;      // i = u*D + d
    int u = i >> 10;
    int d = i & (D_ - 1);
    W1t[i] = (bf16_t)W1[d * U_ + u];
}

// ---------------- K1: phb[b,u] = hidden[b,:]@W2[:,u] + W2_b[u] + W1_b[u] ------
__global__ void k1_proj_h(const float* __restrict__ hidden, const float* __restrict__ W2,
                          const float* __restrict__ W2b, const float* __restrict__ W1b,
                          float* __restrict__ phb) {
    int i = blockIdx.x * 256 + threadIdx.x;      // i = b*U + u
    int b = i >> 9;
    int u = i & (U_ - 1);
    const float* hp = hidden + b * H_;
    float acc = W2b[u] + W1b[u];
#pragma unroll 4
    for (int h = 0; h < H_; ++h)
        acc = fmaf(hp[h], W2[h * U_ + u], acc);
    phb[i] = acc;
}

// ---------------- K2: scores via bf16 WMMA ------------------------------------
// block = 256 threads (8 waves), one block per (b, 64 t-rows)
__global__ void __launch_bounds__(256)
k2_score(const float* __restrict__ feat, const bf16_t* __restrict__ W1t,
         const float* __restrict__ phb, const float* __restrict__ Vw,
         float* __restrict__ scoreWs) {
    __shared__ bf16_t Atile[TB * ASTR];   // 132,096 B
    __shared__ float  swave[8][TB];       // per-wave row partials (deterministic)

    const int tid  = threadIdx.x;
    const int b    = blockIdx.x / (T_ / TB);          // T_/TB = 9
    const int tblk = (blockIdx.x % (T_ / TB)) * TB;

    // ---- stage features tile [64 x 1024] fp32 -> bf16 LDS (coalesced) ----
    const float* fbase = feat + (size_t)(b * T_ + tblk) * D_;
#pragma unroll 4
    for (int i = 0; i < (TB * D_ / 4) / 256; ++i) {   // 64 float4 per thread
        int idx  = tid + i * 256;
        int row  = idx >> 8;                          // 256 float4 per row
        int col4 = (idx & 255) << 2;
        float4 f = *(const float4*)(fbase + (size_t)row * D_ + col4);
        v4bf c;
        c[0] = (bf16_t)f.x; c[1] = (bf16_t)f.y;
        c[2] = (bf16_t)f.z; c[3] = (bf16_t)f.w;
        *(v4bf*)&Atile[row * ASTR + col4] = c;
    }
    ((float*)swave)[tid]       = 0.f;                 // 512 slots, 2 per thread
    ((float*)swave)[tid + 256] = 0.f;
    __syncthreads();

    const int wave = tid >> 5, lane = tid & 31;
    const int lrow = lane & 15, lhi = lane >> 4;      // 16-bit operand layout

    for (int uti = 0; uti < 4; ++uti) {
        const int ut   = wave * 4 + uti;              // wave owns 4 u-tiles
        const int ucol = ut * 16 + lrow;              // this lane's output column
        const float ph = phb[b * U_ + ucol];
        const float vw = Vw[ucol];
        const bf16_t* Bp = W1t + (size_t)ucol * D_ + lhi * 8;    // K-contiguous

        for (int tt = 0; tt < 4; ++tt) {
            const bf16_t* Ap = &Atile[(tt * 16 + lrow) * ASTR + lhi * 8];
            v8f acc = {0.f, 0.f, 0.f, 0.f, 0.f, 0.f, 0.f, 0.f};
#pragma unroll 4
            for (int k0 = 0; k0 < D_; k0 += 32) {
                // A: elems 0-7 -> K=base..base+7 ; elems 8-15 -> K=base+16..base+23
                v8bf alo = *(const v8bf*)(Ap + k0);
                v8bf ahi = *(const v8bf*)(Ap + k0 + 16);
                v16bf Af = __builtin_shufflevector(alo, ahi,
                    0, 1, 2, 3, 4, 5, 6, 7, 8, 9, 10, 11, 12, 13, 14, 15);
                // B: elems 0-7 -> K=base..base+7 ; elems 8-15 -> K=base+16..base+23 of column ucol
                v8bf blo = *(const v8bf*)(Bp + k0);
                v8bf bhi = *(const v8bf*)(Bp + k0 + 16);
                v16bf Bf = __builtin_shufflevector(blo, bhi,
                    0, 1, 2, 3, 4, 5, 6, 7, 8, 9, 10, 11, 12, 13, 14, 15);
                acc = __builtin_amdgcn_wmma_f32_16x16x32_bf16(
                        false, Af, false, Bf, (short)0, acc, false, false);
            }
            // epilogue: score[t] += sum_u tanh(P + ph) * Vw[u]
            // c[r] = row (tt*16 + r + 8*lhi), col ucol; reduce over 16 lanes/row
            float pr[8];
#pragma unroll
            for (int r = 0; r < 8; ++r) {
                float p = fast_tanh(acc[r] + ph) * vw;
                p += __shfl_xor(p, 1, 16);
                p += __shfl_xor(p, 2, 16);
                p += __shfl_xor(p, 4, 16);
                p += __shfl_xor(p, 8, 16);
                pr[r] = p;
            }
            if (lrow == 0) {  // single writer per (wave,row): deterministic
#pragma unroll
                for (int r = 0; r < 8; ++r)
                    swave[wave][tt * 16 + lhi * 8 + r] += pr[r];
            }
        }
    }
    __syncthreads();
    if (tid < TB) {
        float s = 0.f;
#pragma unroll
        for (int w = 0; w < 8; ++w) s += swave[w][tid];
        scoreWs[b * T_ + tblk + tid] = s;
    }
}

// ---------------- K3: softmax over T per batch --------------------------------
__global__ void k3_softmax(const float* __restrict__ scoreWs, float* __restrict__ wts) {
    __shared__ float red[256];
    const int b = blockIdx.x, tid = threadIdx.x;
    const float* sc = scoreWs + b * T_;
    float m = -3.4e38f;
    for (int t = tid; t < T_; t += 256) m = fmaxf(m, sc[t]);
    red[tid] = m; __syncthreads();
    for (int s = 128; s > 0; s >>= 1) {
        if (tid < s) red[tid] = fmaxf(red[tid], red[tid + s]);
        __syncthreads();
    }
    const float M = red[0];
    __syncthreads();
    float sum = 0.f;
    for (int t = tid; t < T_; t += 256) sum += __expf(sc[t] - M);
    red[tid] = sum; __syncthreads();
    for (int s = 128; s > 0; s >>= 1) {
        if (tid < s) red[tid] += red[tid + s];
        __syncthreads();
    }
    const float inv = 1.f / red[0];
    for (int t = tid; t < T_; t += 256)
        wts[b * T_ + t] = __expf(sc[t] - M) * inv;
}

// ---------------- K4: context[b,d] = sum_t w[b,t] * features[b,t,d] -----------
__global__ void k4_context(const float* __restrict__ feat, const float* __restrict__ wts,
                           float* __restrict__ ctx) {
    __shared__ float wsh[T_];
    const int tid = threadIdx.x;
    const int b   = blockIdx.y;
    const int d   = blockIdx.x * 256 + tid;
    for (int t = tid; t < T_; t += 256) wsh[t] = wts[b * T_ + t];
    __syncthreads();
    const float* fb = feat + (size_t)b * T_ * D_ + d;
    float acc = 0.f;
#pragma unroll 4
    for (int t = 0; t < T_; ++t)
        acc = fmaf(wsh[t], fb[(size_t)t * D_], acc);
    ctx[b * D_ + d] = acc;
}

// ---------------- launcher ----------------------------------------------------
extern "C" void kernel_launch(void* const* d_in, const int* in_sizes, int n_in,
                              void* d_out, int out_size, void* d_ws, size_t ws_size,
                              hipStream_t stream) {
    const float* feat   = (const float*)d_in[0];   // [B,T,D]
    const float* hidden = (const float*)d_in[1];   // [B,H]
    const float* W1w    = (const float*)d_in[2];   // [D,U]
    const float* W1b    = (const float*)d_in[3];   // [U]
    const float* W2w    = (const float*)d_in[4];   // [H,U]
    const float* W2b    = (const float*)d_in[5];   // [U]
    const float* Vw     = (const float*)d_in[6];   // [U,1]
    // d_in[7] = V_b: uniform shift of scores -> softmax-invariant, exactly droppable

    char* ws = (char*)d_ws;
    bf16_t* W1t    = (bf16_t*)ws;                                  // 1 MiB
    float*  phb    = (float*)(ws + (1u << 20));                    // 256 KiB
    float*  scoreW = (float*)(ws + (1u << 20) + (256u << 10));     // 288 KiB

    float* ctx = (float*)d_out;                 // context_vector [B,D]
    float* wts = (float*)d_out + B_ * D_;       // attention_weights [B,T,1]

    k0_w1_to_bf16t<<<(D_ * U_) / 256, 256, 0, stream>>>(W1w, W1t);
    k1_proj_h<<<(B_ * U_) / 256, 256, 0, stream>>>(hidden, W2w, W2b, W1b, phb);
    k2_score<<<B_ * (T_ / TB), 256, 0, stream>>>(feat, W1t, phb, Vw, scoreW);
    k3_softmax<<<B_, 256, 0, stream>>>(scoreW, wts);
    k4_context<<<dim3(D_ / 256, B_), 256, 0, stream>>>(feat, wts, ctx);
}
